// JatCore_24249385353343
// MI455X (gfx1250) — matrix operand order repaired and checked by reading, not compiled, not run on hardware
//
#include <hip/hip_runtime.h>
#include <hip/hip_bf16.h>
#include <math.h>

typedef __attribute__((ext_vector_type(16))) _Float16 v16h;
typedef __attribute__((ext_vector_type(8)))  _Float16 v8h;
typedef __attribute__((ext_vector_type(8)))  float    v8f;

#define D_FEAT 64
#define LN_EPS 1e-6f
#define BGRP 24            // padded group stride (halves): 48B, 16B-aligned, conflict-free

__device__ __forceinline__ float swishf(float v) {
    return v / (1.0f + expf(-v));
}

__device__ __forceinline__ void atomicMaxF(float* addr, float val) {
    // signed-int compare matches float compare for >=0; unsigned compare is
    // reversed for negatives. Start value is -inf (0xFF800000).
    if (val >= 0.0f) atomicMax((int*)addr, __float_as_int(val));
    else             atomicMin((unsigned int*)addr, __float_as_uint(val));
}

// ---------------------------------------------------------------------------
// Per-layer init: acc = 0, nmax = -inf, nsum = 0
// ---------------------------------------------------------------------------
__global__ void init_layer_kernel(float* __restrict__ acc, float* __restrict__ nmax,
                                  float* __restrict__ nsum, int Nn) {
    int i = blockIdx.x * blockDim.x + threadIdx.x;
    if (i < Nn * D_FEAT) acc[i] = 0.0f;
    if (i < Nn) { nmax[i] = -INFINITY; nsum[i] = 0.0f; }
}

// ---------------------------------------------------------------------------
// q = x @ Wq, k = x @ Wk  (N x 64 @ 64 x 64), WMMA f16->f32.
// 4 waves per block, each wave computes a 16x64 tile of q and k.
// Weights staged in LDS pre-swizzled into B-fragment order so each lane loads
// its 32-byte fragment with two ds_load_b128 (no scalar u16 gathers).
// Fragment layouts used here (wave32; contiguous k order, not the ISA table order):
//   B (32x16 f16): lane-half = K-half; b[t] = W[kc*32 + half*16 + t][col]
//   -> group g = (kc*2 + half)*64 + col holds those 16 halves contiguously.
// ---------------------------------------------------------------------------
__global__ void qk_wmma_kernel(const float* __restrict__ x,
                               const float* __restrict__ Wq,
                               const float* __restrict__ Wk,
                               float* __restrict__ q, float* __restrict__ k, int Nn) {
    __shared__ _Float16 wq_s[256 * BGRP];   // 12 KB
    __shared__ _Float16 wk_s[256 * BGRP];   // 12 KB
    for (int i = threadIdx.x; i < D_FEAT * D_FEAT; i += blockDim.x) {
        const int kk  = i >> 6;             // K row 0..63
        const int col = i & 63;             // output column
        const int g   = ((kk >> 5) * 2 + ((kk >> 4) & 1)) * 64 + col;
        const int t   = kk & 15;
        wq_s[g * BGRP + t] = (_Float16)Wq[i];
        wk_s[g * BGRP + t] = (_Float16)Wk[i];
    }
    __syncthreads();

    const int wave = threadIdx.x >> 5;
    const int lane = threadIdx.x & 31;
    const int m0   = (blockIdx.x * 4 + wave) * 16;
    if (m0 >= Nn) return;

    const int half = lane >> 4;   // lane-half
    const int lq   = lane & 15;   // row (A) / column (B,C,D) within tile

    v8f zero = {0.f,0.f,0.f,0.f,0.f,0.f,0.f,0.f};
    v8f cq[4], ck[4];
#pragma unroll
    for (int t = 0; t < 4; t++) { cq[t] = zero; ck[t] = zero; }

#pragma unroll
    for (int kc = 0; kc < 2; kc++) {            // K in chunks of 32
        // --- A fragment: 16x32 f16; both lane-halves hold rows M=0..15
        v16h a;
        int arow = m0 + lq; if (arow >= Nn) arow = Nn - 1;
        const float* xr = x + (size_t)arow * D_FEAT;
#pragma unroll
        for (int v = 0; v < 8; v++) {
            int kb = kc * 32 + half * 16 + 2 * v;
            a[2 * v]     = (_Float16)xr[kb];
            a[2 * v + 1] = (_Float16)xr[kb + 1];
        }
#pragma unroll
        for (int nt = 0; nt < 4; nt++) {        // four 16-column output tiles
            const int col = nt * 16 + lq;
            const int g   = (kc * 2 + half) * 64 + col;
            const _Float16* pq = wq_s + g * BGRP;
            const _Float16* pk = wk_s + g * BGRP;
            v8h q_lo = *(const v8h*)(pq);
            v8h q_hi = *(const v8h*)(pq + 8);
            v8h k_lo = *(const v8h*)(pk);
            v8h k_hi = *(const v8h*)(pk + 8);
            v16h bq = __builtin_shufflevector(q_lo, q_hi,
                0,1,2,3,4,5,6,7,8,9,10,11,12,13,14,15);
            v16h bk = __builtin_shufflevector(k_lo, k_hi,
                0,1,2,3,4,5,6,7,8,9,10,11,12,13,14,15);
            cq[nt] = __builtin_amdgcn_wmma_f32_16x16x32_f16(
                false, a, false, bq, (short)0, cq[nt], false, false);
            ck[nt] = __builtin_amdgcn_wmma_f32_16x16x32_f16(
                false, a, false, bk, (short)0, ck[nt], false, false);
        }
    }
    // C/D layout: VGPR i -> M = i + 8*half, N = col
#pragma unroll
    for (int nt = 0; nt < 4; nt++) {
        const int col = nt * 16 + lq;
#pragma unroll
        for (int i = 0; i < 8; i++) {
            int row = m0 + half * 8 + i;
            if (row < Nn) {
                q[(size_t)row * D_FEAT + col] = cq[nt][i];
                k[(size_t)row * D_FEAT + col] = ck[nt][i];
            }
        }
    }
}

// ---------------------------------------------------------------------------
// Phase 1: per-edge logit + segment max (by receiver)
// ---------------------------------------------------------------------------
__global__ void edge_logits_kernel(const float* __restrict__ q, const float* __restrict__ k,
                                   const int* __restrict__ snd, const int* __restrict__ rcv,
                                   const float* __restrict__ edges,
                                   const unsigned char* __restrict__ mask,
                                   const float* __restrict__ WaL,   // 65 floats
                                   float* __restrict__ logit, float* __restrict__ nmax, int Ee) {
    int e = blockIdx.x * blockDim.x + threadIdx.x;
    if (e >= Ee) return;
    const int s = snd[e], r = rcv[e];
    const float4* qs = (const float4*)(q + (size_t)s * D_FEAT);
    const float4* kr = (const float4*)(k + (size_t)r * D_FEAT);
    float acc = 0.0f;
#pragma unroll
    for (int t = 0; t < 16; t++) {
        float4 a = qs[t], b = kr[t];
        acc += swishf(a.x + b.x) * WaL[4 * t + 0];
        acc += swishf(a.y + b.y) * WaL[4 * t + 1];
        acc += swishf(a.z + b.z) * WaL[4 * t + 2];
        acc += swishf(a.w + b.w) * WaL[4 * t + 3];
    }
    acc += edges[e] * WaL[64];                  // ep = edges^1 term
    float lg = mask[e] ? acc : -INFINITY;
    logit[e] = lg;
    atomicMaxF(&nmax[r], lg);
}

// ---------------------------------------------------------------------------
// Phase 2: w = exp(logit - max[r]); segment sum
// ---------------------------------------------------------------------------
__global__ void edge_weights_kernel(const float* __restrict__ logit,
                                    const float* __restrict__ nmax,
                                    const int* __restrict__ rcv,
                                    float* __restrict__ wbuf, float* __restrict__ nsum, int Ee) {
    int e = blockIdx.x * blockDim.x + threadIdx.x;
    if (e >= Ee) return;
    const int r = rcv[e];
    float w = expf(logit[e] - nmax[r]);
    wbuf[e] = w;
    atomicAdd(&nsum[r], w);
}

// ---------------------------------------------------------------------------
// Phase 3: acc[r] += (w/sum[r]) * q[s]
// ---------------------------------------------------------------------------
__global__ void edge_aggregate_kernel(const float* __restrict__ q,
                                      const int* __restrict__ snd, const int* __restrict__ rcv,
                                      const float* __restrict__ wbuf,
                                      const float* __restrict__ nsum,
                                      float* __restrict__ acc, int Ee) {
    int e = blockIdx.x * blockDim.x + threadIdx.x;
    if (e >= Ee) return;
    const int s = snd[e], r = rcv[e];
    const float coef = wbuf[e] / nsum[r];
    const float4* qs = (const float4*)(q + (size_t)s * D_FEAT);
    float* ar = acc + (size_t)r * D_FEAT;
#pragma unroll
    for (int t = 0; t < 16; t++) {
        float4 a = qs[t];
        atomicAdd(ar + 4 * t + 0, a.x * coef);
        atomicAdd(ar + 4 * t + 1, a.y * coef);
        atomicAdd(ar + 4 * t + 2, a.z * coef);
        atomicAdd(ar + 4 * t + 3, a.w * coef);
    }
}

// ---------------------------------------------------------------------------
// Node update: x = LN(swish(acc) + x)  (LN skipped on readout layer).
// One wave per node; each lane owns features lane and lane+32.
// ---------------------------------------------------------------------------
__global__ void node_update_kernel(float* __restrict__ x, const float* __restrict__ acc,
                                   const float* __restrict__ ln_s, const float* __restrict__ ln_b,
                                   int readout, int Nn) {
    const int wave = threadIdx.x >> 5;
    const int lane = threadIdx.x & 31;
    const int n = blockIdx.x * (blockDim.x >> 5) + wave;
    if (n >= Nn) return;
    const size_t base = (size_t)n * D_FEAT;
    float y0 = swishf(acc[base + lane])      + x[base + lane];
    float y1 = swishf(acc[base + lane + 32]) + x[base + lane + 32];
    if (!readout) {
        float red = y0 + y1;
#pragma unroll
        for (int off = 16; off; off >>= 1) red += __shfl_xor(red, off, 32);
        const float m = red * (1.0f / 64.0f);
        float d0 = y0 - m, d1 = y1 - m;
        float vv = d0 * d0 + d1 * d1;
#pragma unroll
        for (int off = 16; off; off >>= 1) vv += __shfl_xor(vv, off, 32);
        const float rinv = rsqrtf(vv * (1.0f / 64.0f) + LN_EPS);
        y0 = d0 * rinv * ln_s[lane]      + ln_b[lane];
        y1 = d1 * rinv * ln_s[lane + 32] + ln_b[lane + 32];
    }
    x[base + lane]      = y0;
    x[base + lane + 32] = y1;
}

// ---------------------------------------------------------------------------
// Readout MLP: 64 -> 64 -> 32 -> 16 -> 16 -> 1, swish(LN(h@W)) each dense,
// final swish(h@Wr4). One wave per node; cross-lane data via shuffles only.
// ---------------------------------------------------------------------------
__global__ void readout_kernel(const float* __restrict__ xbuf,
                               const float* __restrict__ W0, const float* __restrict__ W1,
                               const float* __restrict__ W2, const float* __restrict__ W3,
                               const float* __restrict__ W4,
                               const float* __restrict__ RS0, const float* __restrict__ RB0,
                               const float* __restrict__ RS1, const float* __restrict__ RB1,
                               const float* __restrict__ RS2, const float* __restrict__ RB2,
                               const float* __restrict__ RS3, const float* __restrict__ RB3,
                               float* __restrict__ out, int Nn) {
    __shared__ float w0[4096], w1[2048], w2[512], w3[256], w4[16];
    __shared__ float s0[64], b0[64], s1[32], b1[32], s2[16], b2[16], s3[16], b3[16];
    for (int i = threadIdx.x; i < 4096; i += blockDim.x) w0[i] = W0[i];
    for (int i = threadIdx.x; i < 2048; i += blockDim.x) w1[i] = W1[i];
    for (int i = threadIdx.x; i < 512;  i += blockDim.x) w2[i] = W2[i];
    for (int i = threadIdx.x; i < 256;  i += blockDim.x) w3[i] = W3[i];
    if (threadIdx.x < 16) w4[threadIdx.x] = W4[threadIdx.x];
    if (threadIdx.x < 64) { s0[threadIdx.x] = RS0[threadIdx.x]; b0[threadIdx.x] = RB0[threadIdx.x]; }
    if (threadIdx.x < 32) { s1[threadIdx.x] = RS1[threadIdx.x]; b1[threadIdx.x] = RB1[threadIdx.x]; }
    if (threadIdx.x < 16) {
        s2[threadIdx.x] = RS2[threadIdx.x]; b2[threadIdx.x] = RB2[threadIdx.x];
        s3[threadIdx.x] = RS3[threadIdx.x]; b3[threadIdx.x] = RB3[threadIdx.x];
    }
    __syncthreads();

    const int wave = threadIdx.x >> 5;
    const int lane = threadIdx.x & 31;
    const int n = blockIdx.x * (blockDim.x >> 5) + wave;
    if (n >= Nn) return;

    const float* h = xbuf + (size_t)n * D_FEAT;

    // ---- dense 0: 64 -> 64 (lane owns outputs lane, lane+32)
    float a0 = 0.0f, a1 = 0.0f;
    for (int j = 0; j < 64; j++) {
        const float hv = h[j];
        a0 += hv * w0[j * 64 + lane];
        a1 += hv * w0[j * 64 + lane + 32];
    }
    float red = a0 + a1;
#pragma unroll
    for (int off = 16; off; off >>= 1) red += __shfl_xor(red, off, 32);
    float m = red * (1.0f / 64.0f);
    float d0 = a0 - m, d1 = a1 - m;
    float vv = d0 * d0 + d1 * d1;
#pragma unroll
    for (int off = 16; off; off >>= 1) vv += __shfl_xor(vv, off, 32);
    float rinv = rsqrtf(vv * (1.0f / 64.0f) + LN_EPS);
    const float h0 = swishf(d0 * rinv * s0[lane] + b0[lane]);
    const float h1 = swishf(d1 * rinv * s0[lane + 32] + b0[lane + 32]);

    // ---- dense 1: 64 -> 32 (lane owns output lane)
    float acc = 0.0f;
    for (int j = 0; j < 32; j++) acc += __shfl(h0, j, 32) * w1[j * 32 + lane];
    for (int j = 0; j < 32; j++) acc += __shfl(h1, j, 32) * w1[(j + 32) * 32 + lane];
    red = acc;
#pragma unroll
    for (int off = 16; off; off >>= 1) red += __shfl_xor(red, off, 32);
    m = red * (1.0f / 32.0f);
    float d = acc - m; vv = d * d;
#pragma unroll
    for (int off = 16; off; off >>= 1) vv += __shfl_xor(vv, off, 32);
    rinv = rsqrtf(vv * (1.0f / 32.0f) + LN_EPS);
    const float h2 = swishf(d * rinv * s1[lane] + b1[lane]);

    // ---- dense 2: 32 -> 16 (lanes 16..31 duplicate c = lane&15)
    const int c = lane & 15;
    acc = 0.0f;
    for (int j = 0; j < 32; j++) acc += __shfl(h2, j, 32) * w2[j * 16 + c];
    red = acc;
#pragma unroll
    for (int off = 8; off; off >>= 1) red += __shfl_xor(red, off, 32);  // per-16 segment
    m = red * (1.0f / 16.0f);
    d = acc - m; vv = d * d;
#pragma unroll
    for (int off = 8; off; off >>= 1) vv += __shfl_xor(vv, off, 32);
    rinv = rsqrtf(vv * (1.0f / 16.0f) + LN_EPS);
    const float h3 = swishf(d * rinv * s2[c] + b2[c]);

    // ---- dense 3: 16 -> 16 (lane j holds input feature j; j+16 duplicates)
    acc = 0.0f;
    for (int j = 0; j < 16; j++) acc += __shfl(h3, j, 32) * w3[j * 16 + c];
    red = acc;
#pragma unroll
    for (int off = 8; off; off >>= 1) red += __shfl_xor(red, off, 32);
    m = red * (1.0f / 16.0f);
    d = acc - m; vv = d * d;
#pragma unroll
    for (int off = 8; off; off >>= 1) vv += __shfl_xor(vv, off, 32);
    rinv = rsqrtf(vv * (1.0f / 16.0f) + LN_EPS);
    const float h4 = swishf(d * rinv * s3[c] + b3[c]);

    // ---- final: 16 -> 1
    float o = 0.0f;
    for (int j = 0; j < 16; j++) o += __shfl(h4, j, 32) * w4[j];
    if (lane == 0) out[n] = swishf(o);
}

// ---------------------------------------------------------------------------
// Host-side orchestration
// ---------------------------------------------------------------------------
extern "C" void kernel_launch(void* const* d_in, const int* in_sizes, int n_in,
                              void* d_out, int out_size, void* d_ws, size_t ws_size,
                              hipStream_t stream) {
    const float*         nodes = (const float*)d_in[0];
    const float*         edges = (const float*)d_in[1];
    const int*           snd   = (const int*)d_in[2];
    const int*           rcv   = (const int*)d_in[3];
    const unsigned char* mask  = (const unsigned char*)d_in[4];
    const float*         Wq    = (const float*)d_in[5];   // (4,1,64,1,64)
    const float*         Wk    = (const float*)d_in[6];
    const float*         Wa    = (const float*)d_in[7];   // (4,1,65,1,1)
    const float*         lns   = (const float*)d_in[8];   // (3,64)
    const float*         lnb   = (const float*)d_in[9];
    const float* Wr0 = (const float*)d_in[10];
    const float* rs0 = (const float*)d_in[11]; const float* rb0 = (const float*)d_in[12];
    const float* Wr1 = (const float*)d_in[13];
    const float* rs1 = (const float*)d_in[14]; const float* rb1 = (const float*)d_in[15];
    const float* Wr2 = (const float*)d_in[16];
    const float* rs2 = (const float*)d_in[17]; const float* rb2 = (const float*)d_in[18];
    const float* Wr3 = (const float*)d_in[19];
    const float* rs3 = (const float*)d_in[20]; const float* rb3 = (const float*)d_in[21];
    const float* Wr4 = (const float*)d_in[22];

    const int Nn = in_sizes[0] / D_FEAT;
    const int Ee = in_sizes[2];
    const size_t NF = (size_t)Nn * D_FEAT;

    float* ws   = (float*)d_ws;
    float* xb   = ws;            // N*64
    float* qb   = xb + NF;       // N*64
    float* kb   = qb + NF;       // N*64
    float* accb = kb + NF;       // N*64
    float* nmax = accb + NF;     // N
    float* nsum = nmax + Nn;     // N
    float* lg   = nsum + Nn;     // E
    float* wb   = lg + Ee;       // E

    hipMemcpyAsync(xb, nodes, NF * sizeof(float), hipMemcpyDeviceToDevice, stream);

    const int eBlocks = (Ee + 255) / 256;
    const int iBlocks = ((int)NF + 255) / 256;
    const int qkBlocks = (Nn + 63) / 64;       // 4 waves x 16 rows
    const int nodeBlocks = (Nn + 7) / 8;       // 8 waves per block

    for (int l = 0; l < 4; l++) {
        const int readout = (l == 3) ? 1 : 0;
        init_layer_kernel<<<iBlocks, 256, 0, stream>>>(accb, nmax, nsum, Nn);
        qk_wmma_kernel<<<qkBlocks, 128, 0, stream>>>(
            xb, Wq + (size_t)l * 4096, Wk + (size_t)l * 4096, qb, kb, Nn);
        edge_logits_kernel<<<eBlocks, 256, 0, stream>>>(
            qb, kb, snd, rcv, edges, mask, Wa + (size_t)l * 65, lg, nmax, Ee);
        edge_weights_kernel<<<eBlocks, 256, 0, stream>>>(lg, nmax, rcv, wb, nsum, Ee);
        edge_aggregate_kernel<<<eBlocks, 256, 0, stream>>>(qb, snd, rcv, wb, nsum, accb, Ee);
        const float* lsp = readout ? lns : lns + (size_t)l * 64;
        const float* lbp = readout ? lnb : lnb + (size_t)l * 64;
        node_update_kernel<<<nodeBlocks, 256, 0, stream>>>(xb, accb, lsp, lbp, readout, Nn);
    }

    readout_kernel<<<nodeBlocks, 256, 0, stream>>>(
        xb, Wr0, Wr1, Wr2, Wr3, Wr4,
        rs0, rb0, rs1, rb1, rs2, rb2, rs3, rb3,
        (float*)d_out, Nn);
}
